// SpatialAxis_22728966930538
// MI455X (gfx1250) — compile-verified, matrix operand order checked by reading, not run on hardware
//
#include <hip/hip_runtime.h>
#include <stdint.h>

#define N_TOK 32768
#define DM    256
#define NH    8
#define HD    32
#define PATCH 512
#define NBLK  64
#define MLPH  1024

typedef __attribute__((ext_vector_type(16))) __bf16 v16bf;
typedef __attribute__((ext_vector_type(8)))  float  v8f;
typedef __attribute__((ext_vector_type(4)))  int    v4i;

union FragB { uint4 u[2]; v16bf v; };

// ---- CDNA5 async global->LDS path (ASYNCcnt), guarded so compile never breaks.
#if __has_builtin(__builtin_amdgcn_global_load_async_to_lds_b128)
#define HAVE_ASYNC_LDS 1
#else
#define HAVE_ASYNC_LDS 0
#endif

#if __has_builtin(__builtin_amdgcn_s_wait_asynccnt)
#define WAIT_ASYNC0() __builtin_amdgcn_s_wait_asynccnt(0)
#else
#define WAIT_ASYNC0() asm volatile("s_wait_asynccnt 0" ::: "memory")
#endif

__device__ __forceinline__ void ga2l_b128(const void* g, void* l) {
#if HAVE_ASYNC_LDS
  __builtin_amdgcn_global_load_async_to_lds_b128(
      (__attribute__((address_space(1))) v4i*)g,
      (__attribute__((address_space(3))) v4i*)l, 0, 0);
#else
  *(uint4*)l = *(const uint4*)g;
#endif
}

__device__ __forceinline__ unsigned short f2bf(float f) {
  union { float f; unsigned int u; } c; c.f = f;
  unsigned int u = c.u;
  return (unsigned short)((u + 0x7FFFu + ((u >> 16) & 1u)) >> 16);
}
__device__ __forceinline__ v16bf load_frag(const unsigned short* p0, const unsigned short* p1) {
  FragB f;
  f.u[0] = *(const uint4*)p0;   // elements 0..7   (K = lo .. lo+7)
  f.u[1] = *(const uint4*)p1;   // elements 8..15  (K = lo+16 .. lo+23)
  return f.v;
}

// ---------------------------------------------------------------- min/max per axis
__global__ __launch_bounds__(256)
void minmax_kernel(const float* __restrict__ coords, float* __restrict__ mnmx) {
  __shared__ float smn[3][256];
  __shared__ float smx[3][256];
  int tid = threadIdx.x;
  float mn[3] = {1e30f, 1e30f, 1e30f}, mx[3] = {-1e30f, -1e30f, -1e30f};
  for (int n = tid; n < N_TOK; n += 256)
    for (int a = 0; a < 3; a++) {
      float c = coords[n * 3 + a];
      mn[a] = fminf(mn[a], c);
      mx[a] = fmaxf(mx[a], c);
    }
  for (int a = 0; a < 3; a++) { smn[a][tid] = mn[a]; smx[a][tid] = mx[a]; }
  __syncthreads();
  for (int s = 128; s > 0; s >>= 1) {
    if (tid < s)
      for (int a = 0; a < 3; a++) {
        smn[a][tid] = fminf(smn[a][tid], smn[a][tid + s]);
        smx[a][tid] = fmaxf(smx[a][tid], smx[a][tid + s]);
      }
    __syncthreads();
  }
  if (tid < 3) { mnmx[tid] = smn[tid][0]; mnmx[3 + tid] = smx[tid][0]; }
}

// ---------------------------------------------------------------- morton keys
__device__ __forceinline__ unsigned int part1by2(unsigned int v) {
  v &= 0x3FFu;
  v = (v | (v << 16)) & 0x030000FFu;
  v = (v | (v << 8))  & 0x0300F00Fu;
  v = (v | (v << 4))  & 0x030C30C3u;
  v = (v | (v << 2))  & 0x09249249u;
  return v;
}
__global__ __launch_bounds__(256)
void morton_kernel(const float* __restrict__ coords, const float* __restrict__ mnmx,
                   unsigned long long* __restrict__ keys, int layer) {
  int n = blockIdx.x * 256 + threadIdx.x;
  unsigned int qv[3];
#pragma unroll
  for (int k = 0; k < 3; k++) {
    int a = (k + layer) % 3;                    // np.roll(arange(3), -layer)
    float c = coords[n * 3 + a];
    float f = (c - mnmx[a]) / (mnmx[3 + a] - mnmx[a] + 1e-9f) * 1023.0f;
    int qi = (int)f;                            // trunc like astype(int32)
    qi = qi < 0 ? 0 : (qi > 1023 ? 1023 : qi);
    qv[k] = (unsigned int)qi;
  }
  unsigned int code = part1by2(qv[0]) | (part1by2(qv[1]) << 1) | (part1by2(qv[2]) << 2);
  keys[n] = (((unsigned long long)code) << 32) | (unsigned int)n;
}

// ---------------------------------------------------------------- 32K bitonic sort fully in 320KB LDS
__global__ __launch_bounds__(1024)
void sort_kernel(const unsigned long long* __restrict__ keys_in, int* __restrict__ perm, int shift) {
  extern __shared__ unsigned long long sk[];   // 32768 * 8B = 256 KB (CDNA5 WGP LDS)
  int tid = threadIdx.x;
  for (int i = tid; i < N_TOK; i += 1024) sk[i] = keys_in[i];
  __syncthreads();
  for (int k = 2; k <= N_TOK; k <<= 1)
    for (int j = k >> 1; j > 0; j >>= 1) {
      for (int i = tid; i < N_TOK; i += 1024) {
        int ixj = i ^ j;
        if (ixj > i) {
          bool up = ((i & k) == 0);
          unsigned long long a = sk[i], b = sk[ixj];
          bool sw = up ? (a > b) : (a < b);
          if (sw) { sk[i] = b; sk[ixj] = a; }
        }
      }
      __syncthreads();
    }
  for (int i = tid; i < N_TOK; i += 1024) {
    int pos = shift ? ((i + PATCH / 2) & (N_TOK - 1)) : i;   // jnp.roll(perm, 256)
    perm[pos] = (int)(sk[i] & 0xFFFFFFFFull);
  }
}

// ---------------------------------------------------------------- gather (x + positional projection s = c . w_pos)
__global__ __launch_bounds__(256)
void gather_kernel(const float* __restrict__ xw, const float* __restrict__ coords,
                   const int* __restrict__ perm, const float* __restrict__ wpos, // [3][8]
                   float* __restrict__ xs, float* __restrict__ sp) {
  int j = blockIdx.x;
  int t = perm[j];
  int tid = threadIdx.x;
  xs[(size_t)j * DM + tid] = xw[(size_t)t * DM + tid];
  if (tid < NH) {
    float c0 = coords[t * 3 + 0], c1 = coords[t * 3 + 1], c2 = coords[t * 3 + 2];
    sp[j * NH + tid] = c0 * wpos[0 * NH + tid] + c1 * wpos[1 * NH + tid] + c2 * wpos[2 * NH + tid];
  }
}

__global__ __launch_bounds__(256)
void scatter_kernel(const float* __restrict__ y, const int* __restrict__ perm,
                    float* __restrict__ xw) {
  int j = blockIdx.x;
  int t = perm[j];
  xw[(size_t)t * DM + threadIdx.x] = y[(size_t)j * DM + threadIdx.x];
}

// ---------------------------------------------------------------- LayerNorm, one wave32 per token
__global__ __launch_bounds__(256)
void ln_kernel(const float* __restrict__ in, const float* __restrict__ g,
               const float* __restrict__ b, unsigned short* __restrict__ outB,
               float* __restrict__ outF) {
  int wid = threadIdx.x >> 5, lane = threadIdx.x & 31;
  int tok = blockIdx.x * 8 + wid;
  const float* row = in + (size_t)tok * DM;
  float4 x0 = *(const float4*)(row + lane * 8);
  float4 x1 = *(const float4*)(row + lane * 8 + 4);
  float vals[8] = {x0.x, x0.y, x0.z, x0.w, x1.x, x1.y, x1.z, x1.w};
  float s = 0.f, s2 = 0.f;
#pragma unroll
  for (int e = 0; e < 8; e++) { s += vals[e]; s2 += vals[e] * vals[e]; }
#pragma unroll
  for (int off = 1; off < 32; off <<= 1) {
    s += __shfl_xor(s, off);
    s2 += __shfl_xor(s2, off);
  }
  float mean = s * (1.0f / DM);
  float var = s2 * (1.0f / DM) - mean * mean;
  float r = rsqrtf(var + 1e-5f);
#pragma unroll
  for (int e = 0; e < 8; e++) {
    int col = lane * 8 + e;
    float y = (vals[e] - mean) * r * g[col] + b[col];
    if (outB) outB[(size_t)tok * DM + col] = f2bf(y);
    if (outF) outF[(size_t)tok * DM + col] = y;
  }
}

// ---------------------------------------------------------------- weight convert + transpose f32[K][N] -> bf16[N][K]
__global__ __launch_bounds__(256)
void convw_kernel(const float* __restrict__ w, unsigned short* __restrict__ wt, int K, int N) {
  int idx = blockIdx.x * 256 + threadIdx.x;
  if (idx < K * N) {
    int k = idx / N, n = idx % N;
    wt[(size_t)n * K + k] = f2bf(w[idx]);
  }
}

// ---------------------------------------------------------------- WMMA GEMM: [M,K]bf16 x [N,K]bf16^T
// 128x128 block tile, 8 waves of 32x64, K staged 32 at a time via async LDS loads.
// MODE 0: bf16 out + bias (qkv) | MODE 1: f32 out + bias + residual (proj/fc2)
// MODE 2: bf16 out + bias + gelu (fc1).
template <int MODE>
__global__ __launch_bounds__(256)
void gemm_kernel(const unsigned short* __restrict__ A,   // [M][K] bf16
                 const unsigned short* __restrict__ WT,  // [NOUT][K] bf16 (pre-transposed)
                 const float* __restrict__ bias,         // [NOUT]
                 const float* __restrict__ resid,        // [M][NOUT] f32 (MODE 1)
                 float* __restrict__ outF,               // f32 out (MODE 1)
                 unsigned short* __restrict__ outB,      // bf16 out (MODE 0,2)
                 int K, int NOUT) {
  __shared__ unsigned short At[128 * 40];
  __shared__ unsigned short Bt[128 * 40];
  const int tid = threadIdx.x;
  const int wid = tid >> 5, lane = tid & 31;
  const int m0 = blockIdx.y * 128, n0 = blockIdx.x * 128;
  const int wr0 = (wid & 3) * 32, wc0 = (wid >> 2) * 64;
  const int hl = (lane < 16) ? 0 : 8;
  const int ln16 = lane & 15;

  v8f acc[2][4];
#pragma unroll
  for (int i = 0; i < 2; i++)
#pragma unroll
    for (int j = 0; j < 4; j++) acc[i][j] = {};

  const int row_ld = tid >> 1;
  const int kc_ld = (tid & 1) * 16;
  const unsigned short* gA = A + (size_t)(m0 + row_ld) * K + kc_ld;
  const unsigned short* gB = WT + (size_t)(n0 + row_ld) * K + kc_ld;
  unsigned short* lA = At + row_ld * 40 + kc_ld;
  unsigned short* lB = Bt + row_ld * 40 + kc_ld;

  for (int k0 = 0; k0 < K; k0 += 32) {
    // Stage A/B tiles straight into LDS (GLOBAL_LOAD_ASYNC_TO_LDS_B128, ASYNCcnt).
    ga2l_b128(gA + k0, lA);
    ga2l_b128(gA + k0 + 8, lA + 8);
    ga2l_b128(gB + k0, lB);
    ga2l_b128(gB + k0 + 8, lB + 8);
    if (k0 + 32 < K) __builtin_prefetch(gA + k0 + 32, 0, 1);
#if HAVE_ASYNC_LDS
    WAIT_ASYNC0();
#endif
    __syncthreads();
    v16bf af[2], bw[4];
#pragma unroll
    for (int fa = 0; fa < 2; fa++) {
      int r = wr0 + fa * 16 + ln16;
      af[fa] = load_frag(At + r * 40 + hl, At + r * 40 + hl + 16);
    }
#pragma unroll
    for (int fb = 0; fb < 4; fb++) {
      int c = wc0 + fb * 16 + ln16;
      bw[fb] = load_frag(Bt + c * 40 + hl, Bt + c * 40 + hl + 16);
    }
#pragma unroll
    for (int fa = 0; fa < 2; fa++)
#pragma unroll
      for (int fb = 0; fb < 4; fb++)
        acc[fa][fb] = __builtin_amdgcn_wmma_f32_16x16x32_bf16(
            false, af[fa], false, bw[fb], (short)0, acc[fa][fb], false, false);
    __syncthreads();
  }
  const int rhi = (lane >= 16) ? 8 : 0;
#pragma unroll
  for (int fa = 0; fa < 2; fa++)
#pragma unroll
    for (int fb = 0; fb < 4; fb++) {
      const int col = n0 + wc0 + fb * 16 + ln16;
      const float bv = bias[col];
      const size_t rb = (size_t)(m0 + wr0 + fa * 16 + rhi) * NOUT + col;
#pragma unroll
      for (int v = 0; v < 8; v++) {
        float x = acc[fa][fb][v] + bv;
        const size_t idx = rb + (size_t)v * NOUT;
        if constexpr (MODE == 1) {
          x += resid[idx];
          outF[idx] = x;
        } else if constexpr (MODE == 2) {
          float x3 = x * x * x;
          x = 0.5f * x * (1.0f + tanhf(0.7978845608028654f * (x + 0.044715f * x3)));
          outB[idx] = f2bf(x);
        } else {
          outB[idx] = f2bf(x);
        }
      }
    }
}

// ---------------------------------------------------------------- windowed attention, flash-style, WMMA
// grid (8 heads, 64 patches), 256 threads (8 waves), each wave owns 64 q rows.
// Row sums of exp(P) are accumulated on the matrix pipe (P @ ones-column) so only
// the running max needs lane shuffles.
__global__ __launch_bounds__(256)
void attn_kernel(const unsigned short* __restrict__ qkv,  // [N][768] bf16 (q|k|v)
                 const float* __restrict__ sp_all,        // [N][8]
                 const float* __restrict__ b_pos,         // [8]
                 unsigned short* __restrict__ outB) {     // [N][256] bf16
  extern __shared__ unsigned char smem[];
  unsigned short* kT = (unsigned short*)smem;        // [512][32]  32 KB
  unsigned short* vT = kT + PATCH * HD;              // [32][512]  32 KB
  float* sp = (float*)(vT + HD * PATCH);             // [512]       2 KB
  unsigned short* pst = (unsigned short*)(sp + PATCH); // 8x[16][32] 8 KB
  const int h = blockIdx.x, blk = blockIdx.y;
  const int base = blk * PATCH;
  const int tid = threadIdx.x, wid = tid >> 5, lane = tid & 31;
  const int ln16 = lane & 15, hl = (lane < 16) ? 0 : 8;
  const int rhi = (lane >= 16) ? 8 : 0;
  const float scale = 0.17677669529663687f;  // 1/sqrt(32)

  for (int c = tid; c < PATCH * 4; c += 256) {       // K tile: async b128 per chunk
    int q = c >> 2, ch = (c & 3) * 8;
    ga2l_b128(qkv + (size_t)(base + q) * 768 + DM + h * HD + ch, kT + q * HD + ch);
  }
  for (int e = tid; e < PATCH * HD; e += 256) {      // V^T tile (transpose, scalar)
    int q = e >> 5, d = e & 31;
    vT[d * PATCH + q] = qkv[(size_t)(base + q) * 768 + 2 * DM + h * HD + d];
  }
  for (int q = tid; q < PATCH; q += 256) sp[q] = sp_all[(base + q) * NH + h];
#if HAVE_ASYNC_LDS
  WAIT_ASYNC0();
#endif
  __syncthreads();

  const float bph = b_pos[h];
  unsigned short* myp = pst + wid * 512;             // per-wave [16][32]

  // B-fragment with ones in column 0: (P @ ones)[p][0] = row-sum of P.
  v16bf onesf;
  {
    FragB ob;
    unsigned int w = (ln16 == 0) ? 0x3F803F80u : 0u;  // packed bf16 {1.0, 1.0}
    ob.u[0] = make_uint4(w, w, w, w);
    ob.u[1] = make_uint4(w, w, w, w);
    onesf = ob.v;
  }

  for (int rt = 0; rt < 4; rt++) {
    int pr = wid * 64 + rt * 16;
    v16bf qa;
    {
      const unsigned short* qrow = qkv + (size_t)(base + pr + ln16) * 768 + h * HD;
      qa = load_frag(qrow + hl, qrow + hl + 16);
    }
    float sprow[8];
#pragma unroll
    for (int v = 0; v < 8; v++) sprow[v] = sp[pr + v + rhi];
    float m[8];
    v8f o0 = {}, o1 = {}, lacc = {};
#pragma unroll
    for (int v = 0; v < 8; v++) m[v] = -1e30f;

    for (int j = 0; j < PATCH / 32; j++) {
      int q0 = j * 32;
      v16bf kb0, kb1;
      { const unsigned short* r = kT + (q0 + ln16) * HD;      kb0 = load_frag(r + hl, r + hl + 16); }
      { const unsigned short* r = kT + (q0 + 16 + ln16) * HD; kb1 = load_frag(r + hl, r + hl + 16); }
      v8f z0 = {}, z1 = {};
      v8f s0 = __builtin_amdgcn_wmma_f32_16x16x32_bf16(false, qa, false, kb0, (short)0, z0, false, false);
      v8f s1 = __builtin_amdgcn_wmma_f32_16x16x32_bf16(false, qa, false, kb1, (short)0, z1, false, false);
      float spc0 = sp[q0 + ln16], spc1 = sp[q0 + 16 + ln16];
#pragma unroll
      for (int v = 0; v < 8; v++) {
        float a0 = s0[v] * scale + sprow[v] - spc0 + bph;
        float a1 = s1[v] * scale + sprow[v] - spc1 + bph;
        float cm = fmaxf(a0, a1);
        cm = fmaxf(cm, __shfl_xor(cm, 1));
        cm = fmaxf(cm, __shfl_xor(cm, 2));
        cm = fmaxf(cm, __shfl_xor(cm, 4));
        cm = fmaxf(cm, __shfl_xor(cm, 8));
        float nm = fmaxf(m[v], cm);
        float rs = __expf(m[v] - nm);
        float p0 = __expf(a0 - nm), p1 = __expf(a1 - nm);
        m[v] = nm;
        o0[v] *= rs;
        o1[v] *= rs;
        lacc[v] *= rs;
        int prow = v + rhi;
        myp[prow * 32 + ln16] = f2bf(p0);
        myp[prow * 32 + 16 + ln16] = f2bf(p1);
      }
      // C-layout -> A-layout transpose through per-wave LDS; DS ops from one wave
      // complete in order, make the RAW explicit with the CDNA5 split counter.
      asm volatile("s_wait_dscnt 0" ::: "memory");
      v16bf pa;
      { const unsigned short* r = myp + ln16 * 32; pa = load_frag(r + hl, r + hl + 16); }
      v16bf vb0, vb1;
      { const unsigned short* r = vT + ln16 * PATCH + q0;        vb0 = load_frag(r + hl, r + hl + 16); }
      { const unsigned short* r = vT + (16 + ln16) * PATCH + q0; vb1 = load_frag(r + hl, r + hl + 16); }
      o0 = __builtin_amdgcn_wmma_f32_16x16x32_bf16(false, pa, false, vb0, (short)0, o0, false, false);
      o1 = __builtin_amdgcn_wmma_f32_16x16x32_bf16(false, pa, false, vb1, (short)0, o1, false, false);
      lacc = __builtin_amdgcn_wmma_f32_16x16x32_bf16(false, pa, false, onesf, (short)0, lacc, false, false);
    }
#pragma unroll
    for (int v = 0; v < 8; v++) {
      // Row sum lives in column 0 (lane 0 / lane 16); broadcast to the 16-lane group.
      float lv = __shfl(lacc[v], lane & 16);
      float inv = 1.0f / lv;
      size_t row = (size_t)(base + pr + v + rhi);
      outB[row * DM + h * HD + ln16] = f2bf(o0[v] * inv);
      outB[row * DM + h * HD + 16 + ln16] = f2bf(o1[v] * inv);
    }
  }
}

// ================================================================ host
extern "C" void kernel_launch(void* const* d_in, const int* in_sizes, int n_in,
                              void* d_out, int out_size, void* d_ws, size_t ws_size,
                              hipStream_t stream) {
  const float* x_in   = (const float*)d_in[0];
  const float* coords = (const float*)d_in[1];
  const float* ln1_g  = (const float*)d_in[2];
  const float* ln1_b  = (const float*)d_in[3];
  const float* w_qkv  = (const float*)d_in[4];
  const float* b_qkv  = (const float*)d_in[5];
  const float* w_proj = (const float*)d_in[6];
  const float* b_proj = (const float*)d_in[7];
  const float* w_pos  = (const float*)d_in[8];
  const float* b_pos  = (const float*)d_in[9];
  const float* ln2_g  = (const float*)d_in[10];
  const float* ln2_b  = (const float*)d_in[11];
  const float* w_fc1  = (const float*)d_in[12];
  const float* b_fc1  = (const float*)d_in[13];
  const float* w_fc2  = (const float*)d_in[14];
  const float* b_fc2  = (const float*)d_in[15];
  const float* norm_g = (const float*)d_in[16];
  const float* norm_b = (const float*)d_in[17];
  float* out = (float*)d_out;

  char* ws = (char*)d_ws;
  size_t off = 0;
  auto alloc = [&](size_t bytes) { size_t o = off; off += (bytes + 255) & ~(size_t)255; return o; };
  float* xw   = (float*)(ws + alloc((size_t)N_TOK * DM * 4));   // working x
  float* xs   = (float*)(ws + alloc((size_t)N_TOK * DM * 4));   // permuted x; later fc2-out y
  float* x2   = (float*)(ws + alloc((size_t)N_TOK * DM * 4));   // post-attention residual
  float* sp   = (float*)(ws + alloc((size_t)N_TOK * NH * 4));
  unsigned long long* keys = (unsigned long long*)(ws + alloc((size_t)N_TOK * 8));
  int*   perm = (int*)(ws + alloc((size_t)N_TOK * 4));
  float* mnmx = (float*)(ws + alloc(256));
  unsigned short* hbuf  = (unsigned short*)(ws + alloc((size_t)N_TOK * DM * 2));   // ln1-out, then ln2-out
  unsigned short* attnb = (unsigned short*)(ws + alloc((size_t)N_TOK * DM * 2));
  unsigned short* big   = (unsigned short*)(ws + alloc((size_t)N_TOK * MLPH * 2)); // qkv, then fc1-out
  unsigned short* wtq[2], *wtp[2], *wt1[2], *wt2[2];
  for (int i = 0; i < 2; i++) {
    wtq[i] = (unsigned short*)(ws + alloc((size_t)DM * 3 * DM * 2));
    wtp[i] = (unsigned short*)(ws + alloc((size_t)DM * DM * 2));
    wt1[i] = (unsigned short*)(ws + alloc((size_t)DM * MLPH * 2));
    wt2[i] = (unsigned short*)(ws + alloc((size_t)MLPH * DM * 2));
  }
  float* y = xs;            // fc2 output reuses xs (dead by then)
  unsigned short* qkvb = big;
  unsigned short* fc1o = big;

  (void)hipFuncSetAttribute((const void*)sort_kernel,
                            hipFuncAttributeMaxDynamicSharedMemorySize, N_TOK * 8);
  const int attn_lds = PATCH * HD * 2 + HD * PATCH * 2 + PATCH * 4 + 8 * 512 * 2;
  (void)hipFuncSetAttribute((const void*)attn_kernel,
                            hipFuncAttributeMaxDynamicSharedMemorySize, attn_lds);

  (void)hipMemcpyAsync(xw, x_in, (size_t)N_TOK * DM * 4, hipMemcpyDeviceToDevice, stream);
  minmax_kernel<<<1, 256, 0, stream>>>(coords, mnmx);
  for (int i = 0; i < 2; i++) {
    convw_kernel<<<(DM * 3 * DM + 255) / 256, 256, 0, stream>>>(w_qkv + (size_t)i * DM * 3 * DM, wtq[i], DM, 3 * DM);
    convw_kernel<<<(DM * DM + 255) / 256, 256, 0, stream>>>(w_proj + (size_t)i * DM * DM, wtp[i], DM, DM);
    convw_kernel<<<(DM * MLPH + 255) / 256, 256, 0, stream>>>(w_fc1 + (size_t)i * DM * MLPH, wt1[i], DM, MLPH);
    convw_kernel<<<(MLPH * DM + 255) / 256, 256, 0, stream>>>(w_fc2 + (size_t)i * MLPH * DM, wt2[i], MLPH, DM);
  }

  for (int i = 0; i < 2; i++) {
    morton_kernel<<<N_TOK / 256, 256, 0, stream>>>(coords, mnmx, keys, i);
    sort_kernel<<<1, 1024, N_TOK * 8, stream>>>(keys, perm, (i % 2) ? 1 : 0);
    gather_kernel<<<N_TOK, 256, 0, stream>>>(xw, coords, perm, w_pos + (size_t)i * 3 * NH, xs, sp);
    ln_kernel<<<N_TOK / 8, 256, 0, stream>>>(xs, ln1_g + (size_t)i * DM, ln1_b + (size_t)i * DM, hbuf, nullptr);
    gemm_kernel<0><<<dim3(3 * DM / 128, N_TOK / 128), 256, 0, stream>>>(
        hbuf, wtq[i], b_qkv + (size_t)i * 3 * DM, nullptr, nullptr, qkvb, DM, 3 * DM);
    attn_kernel<<<dim3(NH, NBLK), 256, attn_lds, stream>>>(qkvb, sp, b_pos + (size_t)i * NH, attnb);
    gemm_kernel<1><<<dim3(DM / 128, N_TOK / 128), 256, 0, stream>>>(
        attnb, wtp[i], b_proj + (size_t)i * DM, xs, x2, nullptr, DM, DM);
    ln_kernel<<<N_TOK / 8, 256, 0, stream>>>(x2, ln2_g + (size_t)i * DM, ln2_b + (size_t)i * DM, hbuf, nullptr);
    gemm_kernel<2><<<dim3(MLPH / 128, N_TOK / 128), 256, 0, stream>>>(
        hbuf, wt1[i], b_fc1 + (size_t)i * MLPH, nullptr, nullptr, fc1o, DM, MLPH);
    gemm_kernel<1><<<dim3(DM / 128, N_TOK / 128), 256, 0, stream>>>(
        fc1o, wt2[i], b_fc2 + (size_t)i * DM, x2, y, nullptr, MLPH, DM);
    scatter_kernel<<<N_TOK, 256, 0, stream>>>(y, perm, xw);
  }
  ln_kernel<<<N_TOK / 8, 256, 0, stream>>>(xw, norm_g, norm_b, nullptr, out);
}
